// MultiSingleStreamBlockLoraProcessor_62319975465669
// MI455X (gfx1250) — matrix operand order repaired and checked by reading, not compiled, not run on hardware
//
#include <hip/hip_runtime.h>
#include <cmath>
#include <cstdint>

// ---------------------------------------------------------------------------
// MI455X (gfx1250) fused QKV+LoRA+RMSNorm+RoPE+block-masked attention.
// Heavy math: v_wmma_f32_16x16x32_bf16. GEMM stages tiles in LDS with
// global_load_async_to_lds_b128 (ASYNCcnt) and double buffering.
// ---------------------------------------------------------------------------

typedef __attribute__((ext_vector_type(16))) __bf16 v16bf;
typedef __attribute__((ext_vector_type(8)))  __bf16 v8bf;
typedef __attribute__((ext_vector_type(4)))  __bf16 v4bf;
typedef __attribute__((ext_vector_type(8)))  float  v8f;

namespace cfg {
constexpr int S     = 3072;
constexpr int DIM   = 3072;
constexpr int HEADS = 24;
constexpr int HD    = 128;
constexpr int RANK  = 64;
constexpr int BLK   = 2048;          // rows >= BLK are conditioning rows
constexpr int PST   = 40;            // padded LDS row stride (elements, 80 B)
constexpr float EPS = 1e-6f;
constexpr float LORA_SCALE = 1.0f;   // LORA_W * ALPHA / RANK = 1 * 64/64
}

union Frag16 { v16bf v; v8bf h[2]; };

static __device__ __forceinline__ v8f wmma_bf16(const Frag16& a, const Frag16& b, v8f c) {
  // D = A(16x32 bf16) * B(32x16 bf16) + C(16x16 f32)
  return __builtin_amdgcn_wmma_f32_16x16x32_bf16(false, a.v, false, b.v,
                                                 (short)0, c, false, false);
}
static __device__ __forceinline__ v8f zero8() {
  v8f z = {0.f,0.f,0.f,0.f,0.f,0.f,0.f,0.f};
  return z;
}

// 16-byte async copy global -> LDS (tracked by ASYNCcnt, not LOADcnt).
// lds_off is the byte offset within the workgroup's LDS allocation
// (low 32 bits of a generic pointer to __shared__ memory).
static __device__ __forceinline__ void async_ld16(unsigned lds_off, const void* g) {
  asm volatile("global_load_async_to_lds_b128 %0, %1, off"
               :: "v"(lds_off), "v"((unsigned long long)(uintptr_t)g)
               : "memory");
}

// ---------------------------------------------------------------------------
// fp32 -> bf16 conversion (vectorized, n must be a multiple of 4)
// ---------------------------------------------------------------------------
__global__ void cvt_bf16_kernel(const float* __restrict__ src,
                                __bf16* __restrict__ dst, int n4) {
  int i = blockIdx.x * blockDim.x + threadIdx.x;
  if (i >= n4) return;
  const float4 v = reinterpret_cast<const float4*>(src)[i];
  v4bf o;
  o[0] = (__bf16)v.x; o[1] = (__bf16)v.y; o[2] = (__bf16)v.z; o[3] = (__bf16)v.w;
  reinterpret_cast<v4bf*>(dst)[i] = o;
}

// ---------------------------------------------------------------------------
// C[m][n] = sum_k A[m][k] * W[n][k] + bias[n]   (h @ W.T + b), bf16 WMMA.
// WG = 256 threads = 8 waves. WG tile 128x128, wave tile 32x64 (2x4 of 16x16).
// A/B 128x32 tiles double-buffered in LDS via async global->LDS copies:
// per K-step a WG moves 16 KB for 1.05 MFLOP -> 64 FLOP/B from L2.
// ---------------------------------------------------------------------------
__global__ __launch_bounds__(256)
void gemm_qkv_kernel(const __bf16* __restrict__ A, const __bf16* __restrict__ W,
                     const float* __restrict__ bias, float* __restrict__ C) {
  using namespace cfg;
  __shared__ __bf16 Atile[2][128 * PST];   // 10 KB per buffer
  __shared__ __bf16 Btile[2][128 * PST];   // total 40 KB LDS

  const int tid  = threadIdx.x;
  const int wave = tid >> 5, lane = tid & 31;
  const int l16  = lane >> 4, ln = lane & 15;
  const int m0 = blockIdx.x * 128;
  const int n0 = blockIdx.y * 128;
  const int wm = (wave & 3) * 32;          // wave row offset in WG tile
  const int wn = (wave >> 2) * 64;         // wave col offset in WG tile

  // copy chunk assignment: tile = 128 rows x 4 chunks of 16B = 512 chunks;
  // 256 threads -> 2 chunks of A and 2 of B per thread per tile.
  const int c0 = tid, c1 = tid + 256;
  const int r0 = c0 >> 2, p0 = c0 & 3;
  const int r1 = c1 >> 2, p1 = c1 & 3;

  const unsigned aB[2] = { (unsigned)(uintptr_t)&Atile[0][0],
                           (unsigned)(uintptr_t)&Atile[1][0] };
  const unsigned bB[2] = { (unsigned)(uintptr_t)&Btile[0][0],
                           (unsigned)(uintptr_t)&Btile[1][0] };

  auto issue = [&](int buf, int k0) {
    async_ld16(aB[buf] + (unsigned)(r0 * PST + p0 * 8) * 2,
               A + (size_t)(m0 + r0) * DIM + k0 + p0 * 8);
    async_ld16(aB[buf] + (unsigned)(r1 * PST + p1 * 8) * 2,
               A + (size_t)(m0 + r1) * DIM + k0 + p1 * 8);
    async_ld16(bB[buf] + (unsigned)(r0 * PST + p0 * 8) * 2,
               W + (size_t)(n0 + r0) * DIM + k0 + p0 * 8);
    async_ld16(bB[buf] + (unsigned)(r1 * PST + p1 * 8) * 2,
               W + (size_t)(n0 + r1) * DIM + k0 + p1 * 8);
  };

  v8f acc[2][4];
#pragma unroll
  for (int i = 0; i < 2; ++i)
#pragma unroll
    for (int j = 0; j < 4; ++j) acc[i][j] = zero8();

  issue(0, 0);                              // prime buffer 0
  for (int k0 = 0; k0 < DIM; k0 += 32) {
    const int cur = (k0 >> 5) & 1;
    const bool more = (k0 + 32) < DIM;
    if (more) {
      issue(cur ^ 1, k0 + 32);              // prefetch next tile
      // async loads complete in order: <=4 outstanding means current tile done
      asm volatile("s_wait_asynccnt 0x4" ::: "memory");
    } else {
      asm volatile("s_wait_asynccnt 0x0" ::: "memory");
    }
    __syncthreads();                        // whole tile visible to all waves

    const __bf16* Abuf = &Atile[cur][0];
    const __bf16* Bbuf = &Btile[cur][0];
    Frag16 af[2], bfr[4];
#pragma unroll
    for (int i = 0; i < 2; ++i) {
      const __bf16* ap = Abuf + (wm + i*16 + ln) * PST;
      af[i].h[0] = *reinterpret_cast<const v8bf*>(ap + l16*16);
      af[i].h[1] = *reinterpret_cast<const v8bf*>(ap + l16*16 + 8);
    }
#pragma unroll
    for (int j = 0; j < 4; ++j) {
      const __bf16* bp = Bbuf + (wn + j*16 + ln) * PST + l16*16;
      bfr[j].h[0] = *reinterpret_cast<const v8bf*>(bp);
      bfr[j].h[1] = *reinterpret_cast<const v8bf*>(bp + 8);
    }
#pragma unroll
    for (int i = 0; i < 2; ++i)
#pragma unroll
      for (int j = 0; j < 4; ++j)
        acc[i][j] = wmma_bf16(af[i], bfr[j], acc[i][j]);
    __syncthreads();                        // all reads done before overwrite
  }

#pragma unroll
  for (int i = 0; i < 2; ++i)
#pragma unroll
    for (int j = 0; j < 4; ++j)
#pragma unroll
      for (int r = 0; r < 8; ++r) {
        const int row = m0 + wm + i*16 + r + 8*l16;  // C layout: r + 8*(lane/16)
        const int col = n0 + wn + j*16 + ln;         // col = lane%16
        C[(size_t)row * DIM + col] = acc[i][j][r] + bias[col];
      }
}

// ---------------------------------------------------------------------------
// LoRA down: t[which][r][j] = sum_k h[BLK+r][k] * down[j][k]  (rank 64)
// ---------------------------------------------------------------------------
__global__ void lora_down_kernel(const float* __restrict__ h,
                                 const float* __restrict__ qd,
                                 const float* __restrict__ kd,
                                 const float* __restrict__ vd,
                                 float* __restrict__ t) {
  using namespace cfg;
  const int per = (S - BLK) * RANK;
  const int idx = blockIdx.x * blockDim.x + threadIdx.x;
  if (idx >= 3 * per) return;
  const int which = idx / per;
  const int rem   = idx - which * per;
  const int r = rem >> 6, j = rem & 63;
  const float* d  = (which == 0) ? qd : (which == 1) ? kd : vd;
  const float* hr = h + (size_t)(BLK + r) * DIM;
  const float* dr = d + (size_t)j * DIM;
  float acc = 0.f;
  for (int k = 0; k < DIM; ++k) acc = fmaf(hr[k], dr[k], acc);
  t[idx] = acc;
}

// ---------------------------------------------------------------------------
// LoRA up + add: dst[BLK+r][n] += scale * sum_j t[which][r][j] * up[n][j]
// ---------------------------------------------------------------------------
__global__ void lora_up_kernel(const float* __restrict__ t,
                               const float* __restrict__ qu,
                               const float* __restrict__ ku,
                               const float* __restrict__ vu,
                               float* __restrict__ qf, float* __restrict__ kf,
                               float* __restrict__ vf) {
  using namespace cfg;
  const int per = (S - BLK) * DIM;
  const int idx = blockIdx.x * blockDim.x + threadIdx.x;
  if (idx >= 3 * per) return;
  const int which = idx / per;
  const int rem   = idx - which * per;
  const int r = rem / DIM, n = rem - r * DIM;
  const float* up  = (which == 0) ? qu : (which == 1) ? ku : vu;
  float*       dst = (which == 0) ? qf : (which == 1) ? kf : vf;
  const float* tr = t + (size_t)which * (S - BLK) * RANK + (size_t)r * RANK;
  const float* ur = up + (size_t)n * RANK;
  float acc = 0.f;
#pragma unroll 8
  for (int j = 0; j < RANK; ++j) acc = fmaf(tr[j], ur[j], acc);
  dst[(size_t)(BLK + r) * DIM + n] += acc * LORA_SCALE;
}

// ---------------------------------------------------------------------------
// RMSNorm + RoPE + per-head relayout to bf16.
// wave0 -> q, wave1 -> k (rms+rope), wave2 -> v (transpose: vt[head][d][s]).
// ---------------------------------------------------------------------------
__global__ __launch_bounds__(96)
void rmsrope_kernel(const float* __restrict__ qf, const float* __restrict__ kf,
                    const float* __restrict__ vf,
                    const float* __restrict__ wq, const float* __restrict__ wk,
                    const float* __restrict__ cosT, const float* __restrict__ sinT,
                    __bf16* __restrict__ qh, __bf16* __restrict__ kh,
                    __bf16* __restrict__ vt) {
  using namespace cfg;
  const int s = blockIdx.x, head = blockIdx.y;
  const int wave = threadIdx.x >> 5, lane = threadIdx.x & 31;
  const float* src = ((wave == 0) ? qf : (wave == 1) ? kf : vf)
                     + (size_t)s * DIM + head * HD;
  const int d0 = lane * 4;
  float4 x = *reinterpret_cast<const float4*>(src + d0);
  if (wave < 2) {
    float ss = x.x*x.x + x.y*x.y + x.z*x.z + x.w*x.w;
#pragma unroll
    for (int off = 1; off < 32; off <<= 1) ss += __shfl_xor(ss, off, 32);
    const float inv = rsqrtf(ss * (1.0f / HD) + EPS);
    const float* w = (wave == 0) ? wq : wk;
    const float a0 = x.x * inv * w[d0+0];
    const float a1 = x.y * inv * w[d0+1];
    const float a2 = x.z * inv * w[d0+2];
    const float a3 = x.w * inv * w[d0+3];
    const float* cp = cosT + (size_t)s * HD + d0;
    const float* sp = sinT + (size_t)s * HD + d0;
    // out[2i] = x[2i]*c - x[2i+1]*s ; out[2i+1] = x[2i+1]*c + x[2i]*s
    const float y0 = a0*cp[0] - a1*sp[0];
    const float y1 = a1*cp[1] + a0*sp[1];
    const float y2 = a2*cp[2] - a3*sp[2];
    const float y3 = a3*cp[3] + a2*sp[3];
    __bf16* dst = ((wave == 0) ? qh : kh) + ((size_t)head * S + s) * HD + d0;
    v4bf o;
    o[0] = (__bf16)y0; o[1] = (__bf16)y1; o[2] = (__bf16)y2; o[3] = (__bf16)y3;
    *reinterpret_cast<v4bf*>(dst) = o;
  } else {
    // transposed V so attention P*V B-fragments are contiguous loads
    vt[((size_t)head*HD + d0+0)*S + s] = (__bf16)x.x;
    vt[((size_t)head*HD + d0+1)*S + s] = (__bf16)x.y;
    vt[((size_t)head*HD + d0+2)*S + s] = (__bf16)x.z;
    vt[((size_t)head*HD + d0+3)*S + s] = (__bf16)x.w;
  }
}

// ---------------------------------------------------------------------------
// Flash attention with block mask as a pure key-range restriction.
// WG = 128 threads = 4 waves, each wave owns 16 query rows (independent).
// qh/kh: [head][s][128] bf16, vt: [head][d][s] bf16, out: [s][DIM] f32.
// ---------------------------------------------------------------------------
__global__ __launch_bounds__(128)
void flash_attn_kernel(const __bf16* __restrict__ qh, const __bf16* __restrict__ kh,
                       const __bf16* __restrict__ vt, float* __restrict__ out) {
  using namespace cfg;
  __shared__ __bf16 Plds[4][16 * 32];          // per-wave P staging (16 q x 32 k)
  const int wave = threadIdx.x >> 5, lane = threadIdx.x & 31;
  const int l16 = lane >> 4, ln = lane & 15;
  const int head = blockIdx.y;
  const int q0 = blockIdx.x * 64 + wave * 16;
  const float scale = 0.08838834764831845f;    // 1/sqrt(128)

  // Q fragments (kept in registers for the whole K loop): 4 frags over HD=128
  Frag16 qa[4];
  {
    const __bf16* qp = qh + ((size_t)head * S + q0 + ln) * HD;
#pragma unroll
    for (int t = 0; t < 4; ++t) {
      qa[t].h[0] = *reinterpret_cast<const v8bf*>(qp + t*32 + l16*16);
      qa[t].h[1] = *reinterpret_cast<const v8bf*>(qp + t*32 + l16*16 + 8);
    }
  }

  float m_r[8], l_r[8];
  v8f o[8];
#pragma unroll
  for (int r = 0; r < 8; ++r) { m_r[r] = -1e30f; l_r[r] = 0.f; }
#pragma unroll
  for (int t = 0; t < 8; ++t) o[t] = zero8();

  // image queries see all keys; conditioning queries see only keys >= BLK
  const int kstart = (q0 < BLK) ? 0 : BLK;
  __bf16* P = &Plds[wave][0];

  for (int kc = kstart; kc < S; kc += 32) {
    // ---- S = Q @ K^T for two 16-key tiles -------------------------------
    v8f s0 = zero8(), s1 = zero8();
#pragma unroll
    for (int t = 0; t < 4; ++t) {
      Frag16 kb0, kb1;
      const __bf16* kp = kh + ((size_t)head * S + kc + ln) * HD + t*32 + l16*16;
      kb0.h[0] = *reinterpret_cast<const v8bf*>(kp);
      kb0.h[1] = *reinterpret_cast<const v8bf*>(kp + 8);
      kb1.h[0] = *reinterpret_cast<const v8bf*>(kp + (size_t)16*HD);
      kb1.h[1] = *reinterpret_cast<const v8bf*>(kp + (size_t)16*HD + 8);
      s0 = wmma_bf16(qa[t], kb0, s0);
      s1 = wmma_bf16(qa[t], kb1, s1);
    }
    // ---- online softmax (row stats live per lane-half, 16-lane reductions)
    float alpha[8];
#pragma unroll
    for (int r = 0; r < 8; ++r) {
      float a = s0[r] * scale, b = s1[r] * scale;
      float mx = fmaxf(a, b);
#pragma unroll
      for (int off = 1; off < 16; off <<= 1) mx = fmaxf(mx, __shfl_xor(mx, off, 32));
      const float mnew = fmaxf(m_r[r], mx);
      a = __expf(a - mnew); b = __expf(b - mnew);
      float rs = a + b;
#pragma unroll
      for (int off = 1; off < 16; off <<= 1) rs += __shfl_xor(rs, off, 32);
      alpha[r] = __expf(m_r[r] - mnew);
      l_r[r] = l_r[r] * alpha[r] + rs;
      m_r[r] = mnew;
      const int row = r + 8 * l16;             // C-layout -> LDS row-major P
      P[row * 32 + ln]      = (__bf16)a;
      P[row * 32 + 16 + ln] = (__bf16)b;
    }
#pragma unroll
    for (int t = 0; t < 8; ++t)
#pragma unroll
      for (int r = 0; r < 8; ++r) o[t][r] *= alpha[r];

    asm volatile("s_wait_dscnt 0" ::: "memory");

    // ---- reload P as an A-fragment (16x32 bf16) -------------------------
    Frag16 pa;
    pa.h[0] = *reinterpret_cast<const v8bf*>(P + ln*32 + l16*16);
    pa.h[1] = *reinterpret_cast<const v8bf*>(P + ln*32 + l16*16 + 8);

    // ---- O += P @ V  (V transposed => contiguous B-frag loads) ----------
#pragma unroll
    for (int t = 0; t < 8; ++t) {
      Frag16 vb;
      const __bf16* vp = vt + ((size_t)head*HD + t*16 + ln) * S + kc + l16*16;
      vb.h[0] = *reinterpret_cast<const v8bf*>(vp);
      vb.h[1] = *reinterpret_cast<const v8bf*>(vp + 8);
      o[t] = wmma_bf16(pa, vb, o[t]);
    }
  }

  float inv_l[8];
#pragma unroll
  for (int r = 0; r < 8; ++r) inv_l[r] = 1.0f / l_r[r];
#pragma unroll
  for (int t = 0; t < 8; ++t)
#pragma unroll
    for (int r = 0; r < 8; ++r) {
      const int row = q0 + r + 8*l16;
      const int col = head*HD + t*16 + ln;
      out[(size_t)row * DIM + col] = o[t][r] * inv_l[r];
    }
}

// ---------------------------------------------------------------------------
extern "C" void kernel_launch(void* const* d_in, const int* in_sizes, int n_in,
                              void* d_out, int out_size, void* d_ws, size_t ws_size,
                              hipStream_t stream) {
  using namespace cfg;
  (void)in_sizes; (void)n_in; (void)out_size; (void)ws_size;

  const float* h      = (const float*)d_in[0];
  const float* Wq     = (const float*)d_in[1];
  const float* Wk     = (const float*)d_in[2];
  const float* Wv     = (const float*)d_in[3];
  const float* bq     = (const float*)d_in[4];
  const float* bk     = (const float*)d_in[5];
  const float* bv     = (const float*)d_in[6];
  const float* qd     = (const float*)d_in[7];
  const float* kd     = (const float*)d_in[8];
  const float* vd     = (const float*)d_in[9];
  const float* qu     = (const float*)d_in[10];
  const float* ku     = (const float*)d_in[11];
  const float* vu     = (const float*)d_in[12];
  const float* wq_rms = (const float*)d_in[13];
  const float* wk_rms = (const float*)d_in[14];
  const float* cosT   = (const float*)d_in[15];
  const float* sinT   = (const float*)d_in[16];
  float* out = (float*)d_out;

  // ---- workspace carve-up (~246 MB total) -------------------------------
  char* ws = (char*)d_ws;
  size_t off = 0;
  auto alloc = [&](size_t bytes) -> void* {
    void* p = ws + off;
    off = (off + bytes + 255) & ~(size_t)255;
    return p;
  };
  const size_t NE = (size_t)S * DIM;               // 9,437,184 elements
  __bf16* hb   = (__bf16*)alloc(NE * 2);
  __bf16* wqb  = (__bf16*)alloc(NE * 2);
  __bf16* wkb  = (__bf16*)alloc(NE * 2);
  __bf16* wvb  = (__bf16*)alloc(NE * 2);
  float*  qf   = (float*) alloc(NE * 4);
  float*  kf   = (float*) alloc(NE * 4);
  float*  vf   = (float*) alloc(NE * 4);
  float*  tmid = (float*) alloc((size_t)3 * (S - BLK) * RANK * 4);
  __bf16* qhb  = (__bf16*)alloc(NE * 2);
  __bf16* khb  = (__bf16*)alloc(NE * 2);
  __bf16* vtb  = (__bf16*)alloc(NE * 2);

  // ---- 1) fp32 -> bf16 --------------------------------------------------
  const int n4 = (int)(NE / 4);
  const dim3 cb(256), cg((n4 + 255) / 256);
  cvt_bf16_kernel<<<cg, cb, 0, stream>>>(h,  hb,  n4);
  cvt_bf16_kernel<<<cg, cb, 0, stream>>>(Wq, wqb, n4);
  cvt_bf16_kernel<<<cg, cb, 0, stream>>>(Wk, wkb, n4);
  cvt_bf16_kernel<<<cg, cb, 0, stream>>>(Wv, wvb, n4);

  // ---- 2) QKV projections (WMMA bf16, LDS-staged async double buffer) ---
  const dim3 gg(S / 128, DIM / 128);
  gemm_qkv_kernel<<<gg, 256, 0, stream>>>(hb, wqb, bq, qf);
  gemm_qkv_kernel<<<gg, 256, 0, stream>>>(hb, wkb, bk, kf);
  gemm_qkv_kernel<<<gg, 256, 0, stream>>>(hb, wvb, bv, vf);

  // ---- 3/4) LoRA on conditioning rows -----------------------------------
  const int ndown = 3 * (S - BLK) * RANK;
  lora_down_kernel<<<(ndown + 255) / 256, 256, 0, stream>>>(h, qd, kd, vd, tmid);
  const int nup = 3 * (S - BLK) * DIM;
  lora_up_kernel<<<(nup + 255) / 256, 256, 0, stream>>>(tmid, qu, ku, vu, qf, kf, vf);

  // ---- 5) RMSNorm + RoPE + relayout -------------------------------------
  rmsrope_kernel<<<dim3(S, HEADS), 96, 0, stream>>>(qf, kf, vf, wq_rms, wk_rms,
                                                    cosT, sinT, qhb, khb, vtb);

  // ---- 6) block-masked flash attention (WMMA) ---------------------------
  flash_attn_kernel<<<dim3(S / 64, HEADS), 128, 0, stream>>>(qhb, khb, vtb, out);
}
